// HAMAP_42992622633193
// MI455X (gfx1250) — matrix operand order repaired and checked by reading, not compiled, not run on hardware
//
#include <hip/hip_runtime.h>
#include <hip/hip_bf16.h>

#define U_NUM   100000
#define POI_NUM 120
#define EMB     64
#define HID1    256
#define HID2    128
#define HEADS   6
#define ATT_H   8
#define SEQ     5
#define NEDGE   1600000
#define NHA     (HEADS * ATT_H)   // 48
#define NTILES  (U_NUM / 16)      // 6250 exact

typedef __attribute__((ext_vector_type(16))) _Float16 v16h;
typedef __attribute__((ext_vector_type(8)))  float    v8f;

// ---------------------------------------------------------------------------
// WMMA helper (wave32 16x16x32 f16 -> f32)
// ---------------------------------------------------------------------------
__device__ __forceinline__ v8f wmma_f16(v16h a, v16h b, v8f c) {
    return __builtin_amdgcn_wmma_f32_16x16x32_f16(
        /*neg_a=*/false, a, /*neg_b=*/false, b,
        /*c_mod=*/(short)0, c, /*reuse_a=*/false, /*reuse_b=*/false);
}

// ---------------------------------------------------------------------------
// Swizzled fragment accessors.
// B weights live in global scratch pre-swizzled per fragment:
//   addr = ((nt*nkt + kt)*32 + lane)*16 + elem  (16 f16 = 32B contiguous/lane)
// where the lane's elem e is K row k0 + (lane>>4)*8 + (e&7) + 16*(e>>3), col n0+(lane&15).
// ---------------------------------------------------------------------------
__device__ __forceinline__ v16h load_b_sw(const _Float16* __restrict__ B,
                                          int nkt, int nt, int kt, int lane) {
    return *(const v16h*)(B + ((((nt * nkt + kt) * 32) + lane) << 4));
}

// Activations in LDS in A-fragment layout: [kt][lane][16] (32B contiguous/lane).
__device__ __forceinline__ v16h load_a_sw(const _Float16* A, int kt, int lane) {
    return *(const v16h*)(A + (((kt * 32) + lane) << 4));
}

// A-swizzle address for logical (row, k):
//   kt = k>>5, kk = k&31 ; lane = row + 16*((kk>>3)&1) ; elem = (kk&7) + 8*(kk>>4)
__device__ __forceinline__ int a_sw_idx(int row, int k) {
    int kt = k >> 5, kk = k & 31;
    int lt = row + (((kk >> 3) & 1) << 4);
    int el = (kk & 7) + ((kk >> 4) << 3);
    return ((kt * 32 + lt) << 4) + el;
}

// Finalize a C/D accumulator (rows rb..rb+7, logical col n) into A-swizzled LDS.
__device__ __forceinline__ void finalize_store_a(_Float16* dst, v8f acc, float bv,
                                                 bool relu, int n, int rb) {
    int kt = n >> 5, kk = n & 31;
    int laneOff = ((kk >> 3) & 1) << 4;
    int el = (kk & 7) + ((kk >> 4) << 3);
    _Float16* p = dst + ((kt * 32) << 4) + el;
#pragma unroll
    for (int i = 0; i < 8; ++i) {
        float v = acc[i] + bv;
        if (relu) v = v > 0.0f ? v : 0.0f;
        p[(laneOff + rb + i) << 4] = (_Float16)v;
    }
}

// A fragment directly from global fp32 (convert on load; rows contiguous in K).
__device__ __forceinline__ v16h load_a_gf32(const float* X, int ldk, int row0, int k0, int lane) {
    int row = row0 + (lane & 15);
    int kb  = k0 + ((lane >> 4) << 3);
    const float* p = X + (long)row * ldk + kb;
    v16h a;
#pragma unroll
    for (int i = 0; i < 8; ++i) { a[i] = (_Float16)p[i]; a[i + 8] = (_Float16)p[16 + i]; }
    return a;
}

// ---------------------------------------------------------------------------
// Degree / normalization
// ---------------------------------------------------------------------------
__global__ void k_deg_init(float* deg, int n) {
    int i = blockIdx.x * blockDim.x + threadIdx.x;
    if (i < n) deg[i] = 1.0f;   // self loop
}
__global__ void k_deg_scatter(float* deg, const int* dst, int e) {
    int i = blockIdx.x * blockDim.x + threadIdx.x;
    if (i < e) atomicAdd(&deg[dst[i]], 1.0f);
}
__global__ void k_deg_rsqrt(float* deg, int n) {
    int i = blockIdx.x * blockDim.x + threadIdx.x;
    if (i < n) deg[i] = rsqrtf(deg[i]);
}

// ---------------------------------------------------------------------------
// Weight conversion fp32 -> f16 into swizzled B-fragment layout (+pad to 128).
// regions (f16 elems): [0,4096) gcn1 | [4096,8192) gcn2 | [8192,24576) d1u
//   [24576,57344) d2u | [57344,73728) d3u pad | [73728,81920) d1 pad
// ---------------------------------------------------------------------------
__global__ void k_cvt_weights(_Float16* dst, const float* g1, const float* g2,
                              const float* w1, const float* w2, const float* w3,
                              const float* w4) {
    int i = blockIdx.x * blockDim.x + threadIdx.x;
    if (i >= 81920) return;
    const float* src; int Nreal, nkt, base;
    if (i < 4096)       { src = g1; Nreal = 64;  nkt = 2; base = 0; }
    else if (i < 8192)  { src = g2; Nreal = 64;  nkt = 2; base = 4096; }
    else if (i < 24576) { src = w1; Nreal = 256; nkt = 2; base = 8192; }
    else if (i < 57344) { src = w2; Nreal = 128; nkt = 8; base = 24576; }
    else if (i < 73728) { src = w3; Nreal = 120; nkt = 4; base = 57344; }
    else                { src = w4; Nreal = 120; nkt = 2; base = 73728; }
    int j    = i - base;
    int elem = j & 15, lane = (j >> 4) & 31, f = j >> 9;
    int kt = f % nkt, nt = f / nkt;
    int k  = kt * 32 + ((lane >> 4) << 3) + (elem & 7) + ((elem >> 3) << 4);
    int n  = nt * 16 + (lane & 15);
    float v = (n < Nreal) ? src[k * Nreal + n] : 0.0f;
    dst[i] = (_Float16)v;
}

// ---------------------------------------------------------------------------
// WMMA GEMM: Y[U x 64] = X[U x 64] (fp32) @ W[64 x 64] (swizzled f16).
// ---------------------------------------------------------------------------
__global__ __launch_bounds__(128) void k_gemm64(const float* __restrict__ X,
                                                const _Float16* __restrict__ W,
                                                float* __restrict__ Y) {
    int lane = threadIdx.x & 31;
    int wv   = threadIdx.x >> 5;
    int tile = blockIdx.x * 4 + wv;
    if (tile >= NTILES) return;
    int row0 = tile * 16;
    v16h a0 = load_a_gf32(X, EMB, row0, 0, lane);
    v16h a1 = load_a_gf32(X, EMB, row0, 32, lane);
    int col = lane & 15, rb = (lane >> 4) * 8;
#pragma unroll
    for (int nt = 0; nt < 4; ++nt) {
        v8f acc = {};
        acc = wmma_f16(a0, load_b_sw(W, 2, nt, 0, lane), acc);
        acc = wmma_f16(a1, load_b_sw(W, 2, nt, 1, lane), acc);
#pragma unroll
        for (int i = 0; i < 8; ++i)
            Y[(long)(row0 + rb + i) * EMB + nt * 16 + col] = acc[i];
    }
}

// ---------------------------------------------------------------------------
// GCN aggregation: self loop, edge scatter (atomics), bias(+relu)
// ---------------------------------------------------------------------------
__global__ void k_gcn_self(float* acc, const float* xw, const float* dinv, long n) {
    long i = (long)blockIdx.x * blockDim.x + threadIdx.x;
    if (i >= n) return;
    float d = dinv[i >> 6];
    acc[i] = xw[i] * d * d;
}
__global__ void k_gcn_scatter(float* acc, const float* __restrict__ xw,
                              const float* __restrict__ dinv,
                              const int* __restrict__ src, const int* __restrict__ dst,
                              long nwork) {
    long t = (long)blockIdx.x * blockDim.x + threadIdx.x;
    if (t >= nwork) return;
    long e = t >> 4;
    int  q = (int)(t & 15) * 4;
    int  s = src[e], d = dst[e];
    float nrm = dinv[s] * dinv[d];
    const float4 xs = *(const float4*)(xw + (long)s * EMB + q);
    float* ad = acc + (long)d * EMB + q;
    atomicAdd(ad + 0, xs.x * nrm);
    atomicAdd(ad + 1, xs.y * nrm);
    atomicAdd(ad + 2, xs.z * nrm);
    atomicAdd(ad + 3, xs.w * nrm);
}
__global__ void k_bias_act(float* buf, const float* bias, int relu, long n) {
    long i = (long)blockIdx.x * blockDim.x + threadIdx.x;
    if (i >= n) return;
    float v = buf[i] + bias[i & 63];
    if (relu) v = v > 0.0f ? v : 0.0f;
    buf[i] = v;
}

// ---------------------------------------------------------------------------
// cisum[u,d] = sum_s ci_table2[ids[u,s], d]
// ---------------------------------------------------------------------------
__global__ void k_cisum(float* out, const float* __restrict__ tab,
                        const int* __restrict__ ids, long n) {
    long i = (long)blockIdx.x * blockDim.x + threadIdx.x;
    if (i >= n) return;
    long u = i >> 6; int d = (int)(i & 63);
    const int* id = ids + u * SEQ;
    float s = 0.0f;
#pragma unroll
    for (int k = 0; k < SEQ; ++k) s += tab[(long)id[k] * EMB + d];
    out[i] = s;
}

// ---------------------------------------------------------------------------
// Attention collapsed to 5 coeffs/user (see prior round). 1 thread/user.
// ---------------------------------------------------------------------------
__global__ __launch_bounds__(32) void k_attention(
        float* __restrict__ uef, const float* __restrict__ ci_tab,
        const int* __restrict__ ids_all, const float* __restrict__ ue,
        const float* __restrict__ at_w, const float* __restrict__ at_b,
        const float* __restrict__ d2_w, const float* __restrict__ d2_b) {
    __shared__ float s_aw[EMB * NHA];        // [d][ha]
    __shared__ float s_atb[NHA];
    __shared__ float s_d2w[NHA + 1];
    __shared__ float s_wb[32 * SEQ * NHA];   // per-thread w[s][ha]

    int tid = threadIdx.x;
    for (int i = tid; i < EMB * NHA; i += 32) {
        int d = i / NHA, ha = i % NHA;
        int h = ha >> 3, a = ha & 7;
        s_aw[i] = at_w[h * (EMB * ATT_H) + d * ATT_H + a];
    }
    for (int i = tid; i < NHA; i += 32) s_atb[i] = at_b[i];
    for (int i = tid; i < NHA + 1; i += 32) s_d2w[i] = d2_w[i];
    __syncthreads();

    int u = blockIdx.x * 32 + tid;           // U_NUM % 32 == 0
    const int* ids = ids_all + (long)u * SEQ;
    int idr[SEQ];
    float* wb = s_wb + tid * (SEQ * NHA);

    // phase 1: w = ci @ at_w + at_b
    for (int s = 0; s < SEQ; ++s) {
        idr[s] = ids[s];
        float wacc[NHA];
#pragma unroll
        for (int ha = 0; ha < NHA; ++ha) wacc[ha] = s_atb[ha];
        const float4* cr4 = (const float4*)(ci_tab + (long)idr[s] * EMB);
        for (int d4 = 0; d4 < EMB / 4; ++d4) {
            float4 c = cr4[d4];
            const float* aw = s_aw + d4 * 4 * NHA;
#pragma unroll
            for (int ha = 0; ha < NHA; ++ha) {
                wacc[ha] += c.x * aw[ha];
                wacc[ha] += c.y * aw[NHA + ha];
                wacc[ha] += c.z * aw[2 * NHA + ha];
                wacc[ha] += c.w * aw[3 * NHA + ha];
            }
        }
#pragma unroll
        for (int ha = 0; ha < NHA; ++ha) wb[s * NHA + ha] = wacc[ha];
    }

    // phase 2: softmax over s per (h,a); fold in d2_w
    float csc[SEQ];
#pragma unroll
    for (int s = 0; s < SEQ; ++s) csc[s] = 0.0f;
    for (int ha = 0; ha < NHA; ++ha) {
        float v[SEQ], m = -1e30f;
#pragma unroll
        for (int s = 0; s < SEQ; ++s) { v[s] = tanhf(wb[s * NHA + ha]); m = fmaxf(m, v[s]); }
        float sum = 0.0f;
#pragma unroll
        for (int s = 0; s < SEQ; ++s) { v[s] = expf(v[s] - m); sum += v[s]; }
        float wgt = s_d2w[1 + ha] / sum;
#pragma unroll
        for (int s = 0; s < SEQ; ++s) csc[s] += v[s] * wgt;
    }

    // phase 3: uef[d] = ue[d]*w0 + sum_s ci[s,d]*c[s] + d2b   (float4 vectorized)
    float w0 = s_d2w[0];
    float b0 = d2_b[0];
    const float4* uer4 = (const float4*)(ue + (long)u * EMB);
    float4* out4 = (float4*)(uef + (long)u * EMB);
    for (int d4 = 0; d4 < EMB / 4; ++d4) {
        float4 a = uer4[d4];
        float4 r;
        r.x = a.x * w0 + b0; r.y = a.y * w0 + b0;
        r.z = a.z * w0 + b0; r.w = a.w * w0 + b0;
#pragma unroll
        for (int s = 0; s < SEQ; ++s) {
            float4 c = ((const float4*)(ci_tab + (long)idr[s] * EMB))[d4];
            r.x += c.x * csc[s]; r.y += c.y * csc[s];
            r.z += c.z * csc[s]; r.w += c.w * csc[s];
        }
        out4[d4] = r;
    }
}

// ---------------------------------------------------------------------------
// Fused WMMA MLP: uef(16x64) -> relu 256 -> relu 128 -> 120 ; + cisum@d1_w
// 1 wave per 16-user tile, 2 waves per block. Activations kept A-swizzled in LDS.
// ---------------------------------------------------------------------------
__global__ __launch_bounds__(64) void k_mlp(
        float* __restrict__ out,
        const float* __restrict__ uef, const float* __restrict__ cis,
        const _Float16* __restrict__ w1, const _Float16* __restrict__ w2,
        const _Float16* __restrict__ w3, const _Float16* __restrict__ w4,
        const float* __restrict__ b1, const float* __restrict__ b2,
        const float* __restrict__ b3, const float* __restrict__ b4) {
    __shared__ _Float16 sUef[2][16 * EMB];    // 2 kt-frags, A-swizzled
    __shared__ _Float16 sCis[2][16 * EMB];    // 2 kt-frags
    __shared__ _Float16 sH1[2][16 * HID1];    // 8 kt-frags
    __shared__ _Float16 sH2[2][16 * HID2];    // 4 kt-frags

    int lane = threadIdx.x & 31;
    int wv   = threadIdx.x >> 5;
    int tile = blockIdx.x * 2 + wv;          // grid sized exactly
    int u0   = tile * 16;
    int col  = lane & 15, rb = (lane >> 4) * 8;

    for (int idx = lane; idx < 16 * EMB; idx += 32) {
        int row = idx >> 6, k = idx & 63;
        int a = a_sw_idx(row, k);
        sUef[wv][a] = (_Float16)uef[(long)u0 * EMB + idx];
        sCis[wv][a] = (_Float16)cis[(long)u0 * EMB + idx];
    }
    __syncthreads();

    // ---- layer 1: (16x64)@(64x256) + b1, relu -> sH1 (A frags in regs all 16 N-tiles)
    {
        v16h a0 = load_a_sw(sUef[wv], 0, lane);
        v16h a1 = load_a_sw(sUef[wv], 1, lane);
#pragma unroll
        for (int nt = 0; nt < HID1 / 16; ++nt) {
            v8f acc = {};
            acc = wmma_f16(a0, load_b_sw(w1, 2, nt, 0, lane), acc);
            acc = wmma_f16(a1, load_b_sw(w1, 2, nt, 1, lane), acc);
            finalize_store_a(sH1[wv], acc, b1[nt * 16 + col], true, nt * 16 + col, rb);
        }
    }
    __syncthreads();

    // ---- layer 2: (16x256)@(256x128) + b2, relu -> sH2 (kt-outer, 8 accumulators)
    {
        v8f acc[8];
#pragma unroll
        for (int nt = 0; nt < 8; ++nt) acc[nt] = (v8f){};
#pragma unroll
        for (int kt = 0; kt < HID1 / 32; ++kt) {
            v16h a = load_a_sw(sH1[wv], kt, lane);
#pragma unroll
            for (int nt = 0; nt < 8; ++nt)
                acc[nt] = wmma_f16(a, load_b_sw(w2, 8, nt, kt, lane), acc[nt]);
        }
#pragma unroll
        for (int nt = 0; nt < 8; ++nt)
            finalize_store_a(sH2[wv], acc[nt], b2[nt * 16 + col], true, nt * 16 + col, rb);
    }
    __syncthreads();

    // ---- layer 3 + score2: (16x128)@(128x128p) + (16x64)@(64x128p) -> out
    {
        v8f acc[8];
#pragma unroll
        for (int nt = 0; nt < 8; ++nt) acc[nt] = (v8f){};
#pragma unroll
        for (int kt = 0; kt < HID2 / 32; ++kt) {
            v16h a = load_a_sw(sH2[wv], kt, lane);
#pragma unroll
            for (int nt = 0; nt < 8; ++nt)
                acc[nt] = wmma_f16(a, load_b_sw(w3, 4, nt, kt, lane), acc[nt]);
        }
#pragma unroll
        for (int kt = 0; kt < 2; ++kt) {
            v16h a = load_a_sw(sCis[wv], kt, lane);
#pragma unroll
            for (int nt = 0; nt < 8; ++nt)
                acc[nt] = wmma_f16(a, load_b_sw(w4, 2, nt, kt, lane), acc[nt]);
        }
#pragma unroll
        for (int nt = 0; nt < 8; ++nt) {
            int n = nt * 16 + col;
            if (n < POI_NUM) {
                float bv = b3[n] + (float)SEQ * b4[n];
#pragma unroll
                for (int i = 0; i < 8; ++i)
                    out[(long)(u0 + rb + i) * POI_NUM + n] = acc[nt][i] + bv;
            }
        }
    }
}

// ---------------------------------------------------------------------------
// Host launcher
// ---------------------------------------------------------------------------
extern "C" void kernel_launch(void* const* d_in, const int* in_sizes, int n_in,
                              void* d_out, int out_size, void* d_ws, size_t ws_size,
                              hipStream_t stream) {
    (void)in_sizes; (void)n_in; (void)out_size; (void)ws_size;

    const int*   edge_user = (const int*)d_in[0];
    const int*   u_v_edge  = (const int*)d_in[2];
    const float* user_tab  = (const float*)d_in[4];
    const float* ci_tab    = (const float*)d_in[5];
    const float* ci_tab2   = (const float*)d_in[6];
    const float* gcn1_w    = (const float*)d_in[7];
    const float* gcn1_b    = (const float*)d_in[8];
    const float* gcn2_w    = (const float*)d_in[9];
    const float* gcn2_b    = (const float*)d_in[10];
    const float* at_w      = (const float*)d_in[11];
    const float* at_b      = (const float*)d_in[12];
    const float* d2_w      = (const float*)d_in[13];
    const float* d2_b      = (const float*)d_in[14];
    const float* d1u_w     = (const float*)d_in[15];
    const float* d1u_b     = (const float*)d_in[16];
    const float* d2u_w     = (const float*)d_in[17];
    const float* d2u_b     = (const float*)d_in[18];
    const float* d3u_w     = (const float*)d_in[19];
    const float* d3u_b     = (const float*)d_in[20];
    const float* d1_w      = (const float*)d_in[21];
    const float* d1_b      = (const float*)d_in[22];
    float* out = (float*)d_out;

    const int* ci_ids = edge_user + (long)U_NUM * SEQ;   // edge_user[1]
    const int* e_src  = u_v_edge;
    const int* e_dst  = u_v_edge + NEDGE;

    // carve scratch
    char* base = (char*)d_ws;
    size_t off = 0;
    auto alloc = [&](size_t bytes) -> void* {
        void* p = base + off;
        off = (off + bytes + 255) & ~(size_t)255;
        return p;
    };
    float* dinv = (float*)alloc((size_t)U_NUM * 4);
    float* B0   = (float*)alloc((size_t)U_NUM * EMB * 4);   // xw scratch / uef
    float* B1   = (float*)alloc((size_t)U_NUM * EMB * 4);   // ue layer1 / cisum
    float* B2   = (float*)alloc((size_t)U_NUM * EMB * 4);   // ue final
    _Float16* whalf = (_Float16*)alloc((size_t)81920 * 2);
    _Float16* g1h = whalf;            // 64x64   swizzled
    _Float16* g2h = whalf + 4096;     // 64x64   swizzled
    _Float16* w1h = whalf + 8192;     // 64x256  swizzled
    _Float16* w2h = whalf + 24576;    // 256x128 swizzled
    _Float16* w3p = whalf + 57344;    // 128x128 swizzled (padded)
    _Float16* w4p = whalf + 73728;    // 64x128  swizzled (padded)

    const long nUE = (long)U_NUM * EMB;

    // normalization
    k_deg_init   <<<(U_NUM + 255) / 256, 256, 0, stream>>>(dinv, U_NUM);
    k_deg_scatter<<<(NEDGE + 255) / 256, 256, 0, stream>>>(dinv, e_dst, NEDGE);
    k_deg_rsqrt  <<<(U_NUM + 255) / 256, 256, 0, stream>>>(dinv, U_NUM);

    // weights -> f16 swizzled (+pad)
    k_cvt_weights<<<(81920 + 255) / 256, 256, 0, stream>>>(
        whalf, gcn1_w, gcn2_w, d1u_w, d2u_w, d3u_w, d1_w);

    // GCN layer 1
    k_gemm64<<<(NTILES + 3) / 4, 128, 0, stream>>>(user_tab, g1h, B0);
    k_gcn_self<<<(unsigned)((nUE + 255) / 256), 256, 0, stream>>>(B1, B0, dinv, nUE);
    k_gcn_scatter<<<(unsigned)(((long)NEDGE * 16 + 255) / 256), 256, 0, stream>>>(
        B1, B0, dinv, e_src, e_dst, (long)NEDGE * 16);
    k_bias_act<<<(unsigned)((nUE + 255) / 256), 256, 0, stream>>>(B1, gcn1_b, 1, nUE);

    // GCN layer 2
    k_gemm64<<<(NTILES + 3) / 4, 128, 0, stream>>>(B1, g2h, B0);
    k_gcn_self<<<(unsigned)((nUE + 255) / 256), 256, 0, stream>>>(B2, B0, dinv, nUE);
    k_gcn_scatter<<<(unsigned)(((long)NEDGE * 16 + 255) / 256), 256, 0, stream>>>(
        B2, B0, dinv, e_src, e_dst, (long)NEDGE * 16);
    k_bias_act<<<(unsigned)((nUE + 255) / 256), 256, 0, stream>>>(B2, gcn2_b, 0, nUE);

    // cisum (for score2)
    k_cisum<<<(unsigned)((nUE + 255) / 256), 256, 0, stream>>>(B1, ci_tab2, ci_ids, nUE);

    // attention -> uef (into B0)
    k_attention<<<U_NUM / 32, 32, 0, stream>>>(
        B0, ci_tab, ci_ids, B2, at_w, at_b, d2_w, d2_b);

    // fused MLP + score2 -> out
    k_mlp<<<NTILES / 2, 64, 0, stream>>>(
        out, B0, B1, w1h, w2h, w3p, w4p, d1u_b, d2u_b, d3u_b, d1_b);
}
